// TransformerCN_16312285790932
// MI455X (gfx1250) — matrix operand order repaired and checked by reading, not compiled, not run on hardware
//
#include <hip/hip_runtime.h>
#include <math.h>

typedef __attribute__((ext_vector_type(16))) _Float16 v16h;
typedef __attribute__((ext_vector_type(8)))  _Float16 v8h;
typedef __attribute__((ext_vector_type(8)))  float    v8f;

#define NN 100000
#define EE 500000
#define BG 4096
#define HCC 128
#define RSQRT32 0.1767766952966368811f

// ---------------------------------------------------------------------------
// Weight convert: W f32 [K,128] row-major  ->  Wt f16 [128,K] (n-major)
// ---------------------------------------------------------------------------
__global__ void wcvt_kernel(const float* __restrict__ W, _Float16* __restrict__ Wt, int K) {
    int i = blockIdx.x * blockDim.x + threadIdx.x;
    if (i >= 128 * K) return;
    int n = i / K, k = i - n * K;
    Wt[i] = (_Float16)W[k * 128 + n];
}

// ---------------------------------------------------------------------------
// Fused projection GEMM: act [N,K] f32  x  {Wq,Wk,Wv,Ws} -> q16,k16,v16 (f16),
// s (f32, becomes the scatter accumulator). One 16x16 tile per wave,
// 8 waves/block cover all 128 output columns, grid = N/16 blocks.
// ---------------------------------------------------------------------------
template <int K>
__global__ __launch_bounds__(256) void gemm_qkvs(
    const float* __restrict__ act,
    const _Float16* __restrict__ wtq, const _Float16* __restrict__ wtk,
    const _Float16* __restrict__ wtv, const _Float16* __restrict__ wts,
    const float* __restrict__ bq, const float* __restrict__ bk,
    const float* __restrict__ bv, const float* __restrict__ bs,
    _Float16* __restrict__ q16, _Float16* __restrict__ k16,
    _Float16* __restrict__ v16, float* __restrict__ outS)
{
    constexpr int KS = K / 32;
    const int lane = threadIdx.x & 31;
    const int wave = threadIdx.x >> 5;
    const int r16  = lane & 15;
    const int hi   = lane >> 4;
    const int mtile = blockIdx.x;
    const int row = mtile * 16 + r16;          // A-matrix row for this lane
    const int col = wave * 16 + r16;           // C/D column for this lane

    // ---- A fragments (16x32 f16 per k-step), ISA layout:
    // elem i -> K = kbase + (i/8)*16 + hi*8 + (i%8): two runs of 8 contiguous
    v16h afrag[KS];
    const float* ar = act + (size_t)row * K;
#pragma unroll
    for (int ks = 0; ks < KS; ++ks) {
        const int base = ks * 32 + hi * 8;
        float4 f0 = *(const float4*)(ar + base);
        float4 f1 = *(const float4*)(ar + base + 4);
        float4 f2 = *(const float4*)(ar + base + 16);
        float4 f3 = *(const float4*)(ar + base + 20);
        v16h a;
        a[0]=(_Float16)f0.x;  a[1]=(_Float16)f0.y;  a[2]=(_Float16)f0.z;  a[3]=(_Float16)f0.w;
        a[4]=(_Float16)f1.x;  a[5]=(_Float16)f1.y;  a[6]=(_Float16)f1.z;  a[7]=(_Float16)f1.w;
        a[8]=(_Float16)f2.x;  a[9]=(_Float16)f2.y;  a[10]=(_Float16)f2.z; a[11]=(_Float16)f2.w;
        a[12]=(_Float16)f3.x; a[13]=(_Float16)f3.y; a[14]=(_Float16)f3.z; a[15]=(_Float16)f3.w;
        afrag[ks] = a;
    }

    const _Float16* wmat[4] = {wtq, wtk, wtv, wts};
    const float*    bias[4] = {bq, bk, bv, bs};
    _Float16*       o16[3]  = {q16, k16, v16};

#pragma unroll
    for (int wm = 0; wm < 4; ++wm) {
        v8f acc = {};
        // B fragment: lane covers column `col`, elem i -> K = kbase + (i/8)*16 + hi*8 + (i%8)
        const _Float16* wb = wmat[wm] + (size_t)col * K + hi * 8;
#pragma unroll
        for (int ks = 0; ks < KS; ++ks) {
            v8h b0 = *(const v8h*)(wb + ks * 32);
            v8h b1 = *(const v8h*)(wb + ks * 32 + 16);
            v16h b = __builtin_shufflevector(b0, b1, 0,1,2,3,4,5,6,7,8,9,10,11,12,13,14,15);
            acc = __builtin_amdgcn_wmma_f32_16x16x32_f16(
                false, afrag[ks], false, b, (short)0, acc, false, false);
        }
        const float bb = bias[wm][col];
        if (wm < 3) {
            _Float16* o = o16[wm];
#pragma unroll
            for (int j = 0; j < 8; ++j) {
                int r = mtile * 16 + hi * 8 + j;   // C/D layout: M = j (+8 for hi lanes)
                o[(size_t)r * HCC + col] = (_Float16)(acc[j] + bb);
            }
        } else {
#pragma unroll
            for (int j = 0; j < 8; ++j) {
                int r = mtile * 16 + hi * 8 + j;
                outS[(size_t)r * HCC + col] = acc[j] + bb;  // skip connection x@Ws+bs
            }
        }
    }
}

// ---------------------------------------------------------------------------
// Edge pass A: alpha = dot(q[dst], k[src]+e)/sqrt(32) per head; atomicMax/dst.
// One wave per edge; lane l handles channel h*32+l. e recomputed from ea@We.
// ---------------------------------------------------------------------------
__device__ __forceinline__ unsigned encode_ord(float f) {
    unsigned u = __float_as_uint(f);
    return (u & 0x80000000u) ? ~u : (u | 0x80000000u);
}
__device__ __forceinline__ float decode_ord(unsigned u) {
    return (u & 0x80000000u) ? __uint_as_float(u ^ 0x80000000u) : __uint_as_float(~u);
}

__global__ __launch_bounds__(256) void edge_alpha(
    const int* __restrict__ ei, const float* __restrict__ ea,
    const float* __restrict__ We,
    const _Float16* __restrict__ q16, const _Float16* __restrict__ k16,
    float* __restrict__ alphaw, unsigned int* __restrict__ nodemax)
{
    const int lane = threadIdx.x & 31;
    const int edge = blockIdx.x * 8 + (threadIdx.x >> 5);
    if (edge >= EE) return;
    const int src = ei[edge];
    const int dst = ei[EE + edge];
    const float ea0 = ea[edge*4+0], ea1 = ea[edge*4+1], ea2 = ea[edge*4+2], ea3 = ea[edge*4+3];
    float red[4];
#pragma unroll
    for (int h = 0; h < 4; ++h) {
        const int c = h * 32 + lane;
        float e  = ea0*We[c] + ea1*We[HCC+c] + ea2*We[2*HCC+c] + ea3*We[3*HCC+c];
        float qv = (float)q16[(size_t)dst * HCC + c];
        float kv = (float)k16[(size_t)src * HCC + c] + e;
        float p  = qv * kv;
#pragma unroll
        for (int off = 16; off > 0; off >>= 1) p += __shfl_xor(p, off, 32);
        red[h] = p;
    }
    if (lane < 4) {
        float val = (lane == 0 ? red[0] : lane == 1 ? red[1] : lane == 2 ? red[2] : red[3]) * RSQRT32;
        alphaw[(size_t)edge * 4 + lane] = val;
        atomicMax(&nodemax[(size_t)dst * 4 + lane], encode_ord(val));
    }
}

// ---------------------------------------------------------------------------
// Edge pass B: alpha = exp(alpha - max[dst]); atomicAdd into nodesum.
// ---------------------------------------------------------------------------
__global__ void edge_exp(const int* __restrict__ ei, float* __restrict__ alphaw,
                         const unsigned int* __restrict__ nodemax,
                         float* __restrict__ nodesum)
{
    int i = blockIdx.x * blockDim.x + threadIdx.x;
    if (i >= EE * 4) return;
    int edge = i >> 2, head = i & 3;
    int dst = ei[EE + edge];
    float m  = decode_ord(nodemax[(size_t)dst * 4 + head]);
    float ex = __expf(alphaw[i] - m);
    alphaw[i] = ex;
    atomicAdd(&nodesum[(size_t)dst * 4 + head], ex);
}

// ---------------------------------------------------------------------------
// Edge pass C: msg = (v[src]+e) * alpha/(sum+eps); scatter-add onto out
// (out already holds x@Ws+bs from the GEMM epilogue).
// ---------------------------------------------------------------------------
__global__ __launch_bounds__(256) void edge_scatter(
    const int* __restrict__ ei, const float* __restrict__ ea,
    const float* __restrict__ We, const _Float16* __restrict__ v16,
    const float* __restrict__ alphaw, const float* __restrict__ nodesum,
    float* __restrict__ out)
{
    const int lane = threadIdx.x & 31;
    const int edge = blockIdx.x * 8 + (threadIdx.x >> 5);
    if (edge >= EE) return;
    const int src = ei[edge];
    const int dst = ei[EE + edge];
    const float ea0 = ea[edge*4+0], ea1 = ea[edge*4+1], ea2 = ea[edge*4+2], ea3 = ea[edge*4+3];
#pragma unroll
    for (int h = 0; h < 4; ++h) {
        float w = alphaw[(size_t)edge * 4 + h] / (nodesum[(size_t)dst * 4 + h] + 1e-16f);
        const int c = h * 32 + lane;
        float e = ea0*We[c] + ea1*We[HCC+c] + ea2*We[2*HCC+c] + ea3*We[3*HCC+c];
        float val = ((float)v16[(size_t)src * HCC + c] + e) * w;
        atomicAdd(&out[(size_t)dst * HCC + c], val);
    }
}

__global__ void relu_kernel(float* __restrict__ p, int n) {
    int i = blockIdx.x * blockDim.x + threadIdx.x;
    if (i < n) p[i] = fmaxf(p[i], 0.0f);
}

// ---------------------------------------------------------------------------
// Global mean pool: one wave per node, atomicAdd into pooled[B,128] + count.
// ---------------------------------------------------------------------------
__global__ __launch_bounds__(256) void pool_kernel(
    const float* __restrict__ h, const int* __restrict__ batch,
    float* __restrict__ pooled, float* __restrict__ cnt)
{
    const int lane = threadIdx.x & 31;
    const int node = blockIdx.x * 8 + (threadIdx.x >> 5);
    if (node >= NN) return;
    const int b = batch[node];
#pragma unroll
    for (int j = 0; j < 4; ++j) {
        int c = j * 32 + lane;
        atomicAdd(&pooled[(size_t)b * HCC + c], h[(size_t)node * HCC + c]);
    }
    if (lane == 0) atomicAdd(&cnt[b], 1.0f);
}

// ---------------------------------------------------------------------------
// Per-task head: out[b] = dot(pooled[b]/max(cnt,1), head_W[t]) + head_b[t]
// ---------------------------------------------------------------------------
__global__ __launch_bounds__(256) void head_kernel(
    const float* __restrict__ pooled, const float* __restrict__ cnt,
    const int* __restrict__ rt, const float* __restrict__ hw,
    const float* __restrict__ hb, float* __restrict__ out)
{
    const int lane = threadIdx.x & 31;
    const int b = blockIdx.x * 8 + (threadIdx.x >> 5);
    if (b >= BG) return;
    const int t = rt[b];
    const float inv = 1.0f / fmaxf(cnt[b], 1.0f);
    float s = 0.0f;
#pragma unroll
    for (int j = 0; j < 4; ++j) {
        int c = j * 32 + lane;
        s += pooled[(size_t)b * HCC + c] * inv * hw[(size_t)t * HCC + c];
    }
#pragma unroll
    for (int off = 16; off > 0; off >>= 1) s += __shfl_xor(s, off, 32);
    if (lane == 0) out[b] = s + hb[t];
}

// ---------------------------------------------------------------------------
extern "C" void kernel_launch(void* const* d_in, const int* in_sizes, int n_in,
                              void* d_out, int out_size, void* d_ws, size_t ws_size,
                              hipStream_t stream)
{
    (void)in_sizes; (void)n_in; (void)out_size; (void)ws_size;

    const float* x     = (const float*)d_in[0];
    const int*   ei    = (const int*)d_in[1];
    const float* ea    = (const float*)d_in[2];
    const int*   batch = (const int*)d_in[3];
    const int*   rt    = (const int*)d_in[4];

    // layer weight pointers
    const float* Wq[3] = {(const float*)d_in[5],  (const float*)d_in[14], (const float*)d_in[14] + 128*128};
    const float* bq[3] = {(const float*)d_in[6],  (const float*)d_in[15], (const float*)d_in[15] + 128};
    const float* Wk[3] = {(const float*)d_in[7],  (const float*)d_in[16], (const float*)d_in[16] + 128*128};
    const float* bk[3] = {(const float*)d_in[8],  (const float*)d_in[17], (const float*)d_in[17] + 128};
    const float* Wv[3] = {(const float*)d_in[9],  (const float*)d_in[18], (const float*)d_in[18] + 128*128};
    const float* bv[3] = {(const float*)d_in[10], (const float*)d_in[19], (const float*)d_in[19] + 128};
    const float* We[3] = {(const float*)d_in[11], (const float*)d_in[20], (const float*)d_in[20] + 4*128};
    const float* Ws[3] = {(const float*)d_in[12], (const float*)d_in[21], (const float*)d_in[21] + 128*128};
    const float* bs[3] = {(const float*)d_in[13], (const float*)d_in[22], (const float*)d_in[22] + 128};
    const float* headW = (const float*)d_in[23];
    const float* headB = (const float*)d_in[24];

    // workspace carve-up (256B aligned)
    char* ws = (char*)d_ws;
    size_t off = 0;
    auto carve = [&](size_t bytes) -> char* {
        char* p = ws + off;
        off += (bytes + 255) & ~(size_t)255;
        return p;
    };
    float*     bufA    = (float*)carve((size_t)NN * HCC * 4);
    float*     bufB    = (float*)carve((size_t)NN * HCC * 4);
    _Float16*  q16     = (_Float16*)carve((size_t)NN * HCC * 2);
    _Float16*  k16     = (_Float16*)carve((size_t)NN * HCC * 2);
    _Float16*  v16     = (_Float16*)carve((size_t)NN * HCC * 2);
    float*     alphaw  = (float*)carve((size_t)EE * 4 * 4);
    unsigned*  nodemax = (unsigned*)carve((size_t)NN * 4 * 4);
    float*     nodesum = (float*)carve((size_t)NN * 4 * 4);
    float*     pooled  = (float*)carve((size_t)BG * HCC * 4);
    float*     cnt     = (float*)carve((size_t)BG * 4);
    _Float16*  wtq[3]; _Float16* wtk[3]; _Float16* wtv[3]; _Float16* wts[3];
    const int  Kdim[3] = {64, 128, 128};
    for (int l = 0; l < 3; ++l) {
        wtq[l] = (_Float16*)carve((size_t)128 * Kdim[l] * 2);
        wtk[l] = (_Float16*)carve((size_t)128 * Kdim[l] * 2);
        wtv[l] = (_Float16*)carve((size_t)128 * Kdim[l] * 2);
        wts[l] = (_Float16*)carve((size_t)128 * Kdim[l] * 2);
    }

    // convert + transpose weights to f16 B-fragment-friendly layout
    for (int l = 0; l < 3; ++l) {
        int K = Kdim[l];
        int grid = (128 * K + 255) / 256;
        wcvt_kernel<<<grid, 256, 0, stream>>>(Wq[l], wtq[l], K);
        wcvt_kernel<<<grid, 256, 0, stream>>>(Wk[l], wtk[l], K);
        wcvt_kernel<<<grid, 256, 0, stream>>>(Wv[l], wtv[l], K);
        wcvt_kernel<<<grid, 256, 0, stream>>>(Ws[l], wts[l], K);
    }

    const float* acts[3] = {x, bufA, bufB};
    float*       outs[3] = {bufA, bufB, bufA};

    for (int l = 0; l < 3; ++l) {
        hipMemsetAsync(nodemax, 0, (size_t)NN * 4 * 4, stream);
        hipMemsetAsync(nodesum, 0, (size_t)NN * 4 * 4, stream);

        if (l == 0) {
            gemm_qkvs<64><<<NN / 16, 256, 0, stream>>>(
                acts[l], wtq[l], wtk[l], wtv[l], wts[l],
                bq[l], bk[l], bv[l], bs[l], q16, k16, v16, outs[l]);
        } else {
            gemm_qkvs<128><<<NN / 16, 256, 0, stream>>>(
                acts[l], wtq[l], wtk[l], wtv[l], wts[l],
                bq[l], bk[l], bv[l], bs[l], q16, k16, v16, outs[l]);
        }

        edge_alpha<<<EE / 8, 256, 0, stream>>>(ei, ea, We[l], q16, k16, alphaw, nodemax);
        edge_exp<<<(EE * 4 + 255) / 256, 256, 0, stream>>>(ei, alphaw, nodemax, nodesum);
        edge_scatter<<<EE / 8, 256, 0, stream>>>(ei, ea, We[l], v16, alphaw, nodesum, outs[l]);

        if (l < 2) relu_kernel<<<(NN * HCC + 255) / 256, 256, 0, stream>>>(outs[l], NN * HCC);
    }

    hipMemsetAsync(pooled, 0, (size_t)BG * HCC * 4, stream);
    hipMemsetAsync(cnt, 0, (size_t)BG * 4, stream);
    pool_kernel<<<NN / 8, 256, 0, stream>>>(bufA, batch, pooled, cnt);
    head_kernel<<<BG / 8, 256, 0, stream>>>(pooled, cnt, rt, headW, headB, (float*)d_out);
}
